// MSFE_86620900425754
// MI455X (gfx1250) — matrix operand order repaired and checked by reading, not compiled, not run on hardware
//
#include <hip/hip_runtime.h>

// ---------------------------------------------------------------------------
// MSFE block for MI455X (gfx1250, wave32, WMMA 16x16x32 f16).
// All GEMMs are NT-form: C[m,n] = sum_k A[m*lda+k] * B[n*ldb+k]
// which matches both torch [out,in] weights and our transposed activations,
// giving contiguous per-lane WMMA fragment loads per the CDNA5 ISA layouts.
// ---------------------------------------------------------------------------

#define CDIM   256
#define HWD    64
#define HW     4096      // 64*64
#define BATCH  8
#define NTOK   32768     // 8*4096
#define CH     1024
#define EPS_DYT  1e-5f
#define EPS_ATTN 1e-5f

typedef _Float16 v16h __attribute__((ext_vector_type(16)));
typedef _Float16 v8h  __attribute__((ext_vector_type(8)));
typedef float    v8f  __attribute__((ext_vector_type(8)));

// A-fragment (16x32 f16, ISA 7.12.2): lane<16 -> row=lane, K = k0+[0..7],[16..23]
//                                     lane>=16 -> row=lane-16, K = k0+[8..15],[24..31]
static __device__ __forceinline__ v16h load_a_frag(const _Float16* A, int lda,
                                                   int m0, int k0, int lane) {
  const int m  = m0 + (lane & 15);
  const int kb = k0 + ((lane >> 4) << 3);
  const _Float16* p = A + (size_t)m * lda + kb;
  v8h lo = *(const v8h*)(p);
  v8h hi = *(const v8h*)(p + 16);
  v16h r;
#pragma unroll
  for (int i = 0; i < 8; ++i) { r[i] = lo[i]; r[8 + i] = hi[i]; }
  return r;
}

// B-fragment (32x16 f16): lane<16 -> col=lane, K = k0+[0..7],[16..23]; lane>=16 -> K = k0+[8..15],[24..31]
// B stored as [Ncols][K] row-major (torch [out,in] weight layout / our transposed acts).
static __device__ __forceinline__ v16h load_b_frag(const _Float16* Bm, int ldb,
                                                   int n0, int k0, int lane) {
  const int n  = n0 + (lane & 15);
  const int kb = k0 + ((lane >> 4) << 3);
  const _Float16* p = Bm + (size_t)n * ldb + kb;
  v8h lo = *(const v8h*)(p);
  v8h hi = *(const v8h*)(p + 16);
  v16h r;
#pragma unroll
  for (int i = 0; i < 8; ++i) { r[i] = lo[i]; r[8 + i] = hi[i]; }
  return r;
}

static __device__ __forceinline__ v8f wmma_f16(v16h a, v16h b, v8f c) {
  return __builtin_amdgcn_wmma_f32_16x16x32_f16(false, a, false, b, (short)0, c,
                                                false, false);
}

// one wave computes a 16x64 patch = 4 accumulators, sharing the A fragment
static __device__ __forceinline__ void gemm4(const _Float16* A, int lda,
                                             const _Float16* Bm, int ldb,
                                             int m0, int n0, int K, int lane,
                                             v8f acc[4]) {
  for (int k0 = 0; k0 < K; k0 += 32) {
    v16h a = load_a_frag(A, lda, m0, k0, lane);
#pragma unroll
    for (int t = 0; t < 4; ++t) {
      v16h b = load_b_frag(Bm, ldb, n0 + (t << 4), k0, lane);
      acc[t] = wmma_f16(a, b, acc[t]);
    }
  }
}

// C/D element r of v8f (ISA 7.12.2): row = r + 8*(lane>=16), col = lane&15
#define EPILOGUE_BEGIN()                                     \
  const int half = lane >> 4;                                \
  const int nl   = lane & 15;                                \
  _Pragma("unroll") for (int t = 0; t < 4; ++t)              \
  _Pragma("unroll") for (int r = 0; r < 8; ++r) {            \
    const int row = m0 + r + (half << 3);                    \
    const int col = n0 + (t << 4) + nl;                      \
    float v = acc[t][r];
#define EPILOGUE_END() }

// ------------------------------- VALU kernels ------------------------------

__global__ __launch_bounds__(256) void k_cvt(const float* s, _Float16* d, int n) {
  int i = blockIdx.x * 256 + threadIdx.x;
  if (i < n) d[i] = (_Float16)s[i];
}

__global__ __launch_bounds__(256) void k_pos(float* pos) {
  const int hw = blockIdx.x, c = threadIdx.x;
  const int x = hw & 63, y = hw >> 6;
  const int idx = c & 63;
  const float omega = __expf(-logf(10000.f) * (float)idx * (1.f / 64.f));
  const float coord = (c < 128) ? (float)x : (float)y;
  const float ph = coord * omega;
  pos[hw * CDIM + c] = ((c >> 6) & 1) ? cosf(ph) : sinf(ph);
}

// h = silu(x_bchw @ W^T) in NHWC, plus f16 copy for the WMMA stages
__global__ __launch_bounds__(256) void k_inproj(const float* __restrict__ x,
                                                const float* __restrict__ w,
                                                float* h32, _Float16* h16) {
  __shared__ float xs[64];
  const int tok = blockIdx.x, c = threadIdx.x;
  const int b = tok >> 12, hw = tok & 4095;
  if (c < 64) xs[c] = x[(size_t)(b * 64 + c) * HW + hw];
  __syncthreads();
  float acc = 0.f;
#pragma unroll 4
  for (int k = 0; k < 64; ++k) acc += xs[k] * w[c * 64 + k];
  const float s = acc / (1.f + __expf(-acc));
  const size_t i = (size_t)tok * CDIM + c;
  h32[i] = s;
  h16[i] = (_Float16)s;
}

// psi = sigmoid(depthwise 3x3(h)), C=256, NHWC
__global__ __launch_bounds__(256) void k_psi(const float* __restrict__ h,
                                             const float* __restrict__ w,
                                             const float* __restrict__ bias,
                                             float* psi) {
  const int tok = blockIdx.x, c = threadIdx.x;
  const int b = tok >> 12, hw = tok & 4095, y = hw >> 6, x = hw & 63;
  float acc = bias[c];
#pragma unroll
  for (int dy = -1; dy <= 1; ++dy) {
    const int yy = y + dy;
    if (yy < 0 || yy > 63) continue;
#pragma unroll
    for (int dx = -1; dx <= 1; ++dx) {
      const int xx = x + dx;
      if (xx < 0 || xx > 63) continue;
      acc += h[((size_t)(b << 12) + yy * 64 + xx) * CDIM + c] *
             w[c * 9 + (dy + 1) * 3 + (dx + 1)];
    }
  }
  psi[(size_t)tok * CDIM + c] = 1.f / (1.f + __expf(-acc));
}

// ksum[b][c] = sum_n phiKT[b][c][n]
__global__ __launch_bounds__(256) void k_ksum(const _Float16* __restrict__ phiKT,
                                              float* ksum) {
  const int wid = blockIdx.x * 8 + (threadIdx.x >> 5);  // (b*256+c)
  const int lane = threadIdx.x & 31;
  const _Float16* p = phiKT + (size_t)wid * HW;
  float s = 0.f;
  for (int i = lane; i < HW; i += 32) s += (float)p[i];
  for (int o = 16; o; o >>= 1) s += __shfl_down(s, o, 32);
  if (!lane) ksum[wid] = s;
}

// den[tok] = phiQ[tok,:] . ksum[b,:] + eps
__global__ __launch_bounds__(256) void k_den(const _Float16* __restrict__ phiQ,
                                             const float* __restrict__ ksum,
                                             float* den) {
  const int tok = blockIdx.x * 8 + (threadIdx.x >> 5);
  const int lane = threadIdx.x & 31;
  const _Float16* q = phiQ + (size_t)tok * CDIM;
  const float* ks = ksum + ((tok >> 12) << 8);
  float s = 0.f;
#pragma unroll
  for (int j = 0; j < 8; ++j) {
    const int c = lane + (j << 5);
    s += (float)q[c] * ks[c];
  }
  for (int o = 16; o; o >>= 1) s += __shfl_down(s, o, 32);
  if (!lane) den[tok] = s + EPS_ATTN;
}

// DyT (tanh norm, unbiased std) + SE pooling accumulation. In-place safe.
__global__ __launch_bounds__(256) void k_dyt_pool(const float* __restrict__ X,
                                                  const float* __restrict__ R,
                                                  const float* __restrict__ g,
                                                  const float* __restrict__ bt,
                                                  float* D, float* pool) {
  __shared__ float red[256];
  const int tok = blockIdx.x, c = threadIdx.x;
  const size_t i = (size_t)tok * CDIM + c;
  const float t = X[i] + (R ? R[i] : 0.f);
  red[c] = t;
  __syncthreads();
  for (int s2 = 128; s2 > 0; s2 >>= 1) { if (c < s2) red[c] += red[c + s2]; __syncthreads(); }
  const float mu = red[0] * (1.f / 256.f);
  __syncthreads();
  const float d0 = t - mu;
  red[c] = d0 * d0;
  __syncthreads();
  for (int s2 = 128; s2 > 0; s2 >>= 1) { if (c < s2) red[c] += red[c + s2]; __syncthreads(); }
  const float sd = sqrtf(red[0] * (1.f / 255.f)) + EPS_DYT;  // ddof=1
  const float dv = tanhf(g[c] * d0 / sd + bt[c]);
  D[i] = dv;
  atomicAdd(&pool[((tok >> 12) << 8) + c], dv);
}

// SE gate weights: s = sigmoid(avgpool @ m^T + b)
__global__ __launch_bounds__(256) void k_mona(const float* __restrict__ pool,
                                              const float* __restrict__ mw,
                                              const float* __restrict__ mb,
                                              float* s) {
  __shared__ float avg[256];
  const int b = blockIdx.x, o = threadIdx.x;
  avg[o] = pool[(b << 8) + o] * (1.f / 4096.f);
  __syncthreads();
  float acc = mb[o];
  for (int c = 0; c < 256; ++c) acc += avg[c] * mw[o * 256 + c];
  s[(b << 8) + o] = 1.f / (1.f + __expf(-acc));
}

__global__ __launch_bounds__(256) void k_gate(const float* __restrict__ D,
                                              const float* __restrict__ s,
                                              float* o32, _Float16* o16) {
  const int tok = blockIdx.x, c = threadIdx.x;
  const size_t i = (size_t)tok * CDIM + c;
  const float v = D[i] * s[((tok >> 12) << 8) + c];
  if (o32) o32[i] = v;
  o16[i] = (_Float16)v;
}

// depthwise 3x3 (C=1024, one image) + tanh-approx GELU -> f16
__global__ __launch_bounds__(256) void k_dwgelu(const float* __restrict__ hE,
                                                const float* __restrict__ w,
                                                const float* __restrict__ bias,
                                                _Float16* g16) {
  const int hw = blockIdx.x, y = hw >> 6, x = hw & 63;
  for (int c = threadIdx.x; c < CH; c += 256) {
    float acc = bias[c];
#pragma unroll
    for (int dy = -1; dy <= 1; ++dy) {
      const int yy = y + dy;
      if (yy < 0 || yy > 63) continue;
#pragma unroll
      for (int dx = -1; dx <= 1; ++dx) {
        const int xx = x + dx;
        if (xx < 0 || xx > 63) continue;
        acc += hE[(size_t)(yy * 64 + xx) * CH + c] * w[c * 9 + (dy + 1) * 3 + (dx + 1)];
      }
    }
    const float u = acc;
    const float th = tanhf(0.7978845608028654f * (u + 0.044715f * u * u * u));
    g16[(size_t)hw * CH + c] = (_Float16)(0.5f * u * (1.f + th));
  }
}

// ------------------------------- WMMA kernels ------------------------------

// Q/K projection: phi = relu(psi*(x@W^T + b + pos))^2 ; K stores transposed [b][c][hw]
__global__ __launch_bounds__(128) void k_gemm_phi(const _Float16* __restrict__ A,
                                                  const _Float16* __restrict__ W,
                                                  const float* __restrict__ bias,
                                                  const float* __restrict__ pos,
                                                  const float* __restrict__ psi,
                                                  _Float16* out, int transposed) {
  const int lane = threadIdx.x & 31, wave = threadIdx.x >> 5;
  const int m0 = blockIdx.x << 4, n0 = wave << 6;
  v8f acc[4] = {};
  gemm4(A, CDIM, W, CDIM, m0, n0, CDIM, lane, acc);
  EPILOGUE_BEGIN()
    const int b = row >> 12, hw = row & 4095;
    v += bias[col] + pos[hw * CDIM + col];
    v *= psi[(size_t)row * CDIM + col];
    v = fmaxf(v, 0.f);
    v *= v;
    if (transposed)
      out[((size_t)((b << 8) + col) << 12) + hw] = (_Float16)v;
    else
      out[(size_t)row * CDIM + col] = (_Float16)v;
  EPILOGUE_END()
}

// V projection, stored transposed VT[b][c][hw]
__global__ __launch_bounds__(128) void k_gemm_v(const _Float16* __restrict__ A,
                                                const _Float16* __restrict__ W,
                                                const float* __restrict__ bias,
                                                _Float16* outT) {
  const int lane = threadIdx.x & 31, wave = threadIdx.x >> 5;
  const int m0 = blockIdx.x << 4, n0 = wave << 6;
  v8f acc[4] = {};
  gemm4(A, CDIM, W, CDIM, m0, n0, CDIM, lane, acc);
  EPILOGUE_BEGIN()
    const int b = row >> 12, hw = row & 4095;
    outT[((size_t)((b << 8) + col) << 12) + hw] = (_Float16)(v + bias[col]);
  EPILOGUE_END()
}

// kvT[b][d][c] = sum_n VT[b][d][n] * phiKT[b][c][n]   (== kv[c,d])
__global__ __launch_bounds__(128) void k_gemm_kv(const _Float16* __restrict__ VT,
                                                 const _Float16* __restrict__ phiKT,
                                                 _Float16* kvT) {
  const int lane = threadIdx.x & 31, wave = threadIdx.x >> 5;
  const int b = blockIdx.y;
  const int m0 = blockIdx.x << 4, n0 = wave << 6;
  const _Float16* A  = VT    + ((size_t)b << 20);  // 256*4096
  const _Float16* Bm = phiKT + ((size_t)b << 20);
  v8f acc[4] = {};
  gemm4(A, HW, Bm, HW, m0, n0, HW, lane, acc);
  EPILOGUE_BEGIN()
    kvT[((size_t)b << 16) + row * CDIM + col] = (_Float16)v;
  EPILOGUE_END()
}

// attn[n][d] = (phiQ[n,:] . kvT[d,:]) / den[n]
__global__ __launch_bounds__(128) void k_gemm_num(const _Float16* __restrict__ phiQ,
                                                  const _Float16* __restrict__ kvT,
                                                  const float* __restrict__ den,
                                                  _Float16* attn) {
  const int lane = threadIdx.x & 31, wave = threadIdx.x >> 5;
  const int m0 = blockIdx.x << 4, n0 = wave << 6;
  const _Float16* Bm = kvT + ((size_t)(m0 >> 12) << 16);
  v8f acc[4] = {};
  gemm4(phiQ, CDIM, Bm, CDIM, m0, n0, CDIM, lane, acc);
  EPILOGUE_BEGIN()
    attn[(size_t)row * CDIM + col] = (_Float16)(v / den[row]);
  EPILOGUE_END()
}

// O projection -> f32 a
__global__ __launch_bounds__(128) void k_gemm_o(const _Float16* __restrict__ A,
                                                const _Float16* __restrict__ W,
                                                const float* __restrict__ bias,
                                                float* out) {
  const int lane = threadIdx.x & 31, wave = threadIdx.x >> 5;
  const int m0 = blockIdx.x << 4, n0 = wave << 6;
  v8f acc[4] = {};
  gemm4(A, CDIM, W, CDIM, m0, n0, CDIM, lane, acc);
  EPILOGUE_BEGIN()
    out[(size_t)row * CDIM + col] = v + bias[col];
  EPILOGUE_END()
}

// EDFFN expand 256 -> 1024 (one image): hE = U1 @ We^T + be
__global__ __launch_bounds__(128) void k_gemm_we(const _Float16* __restrict__ A,
                                                 const _Float16* __restrict__ W,
                                                 const float* __restrict__ bias,
                                                 float* hE) {
  const int lane = threadIdx.x & 31, wave = threadIdx.x >> 5;
  const int m0 = blockIdx.x << 4;
  const int n0 = (blockIdx.y << 8) + (wave << 6);
  v8f acc[4] = {};
  gemm4(A, CDIM, W, CDIM, m0, n0, CDIM, lane, acc);
  EPILOGUE_BEGIN()
    hE[(size_t)row * CH + col] = v + bias[col];
  EPILOGUE_END()
}

// EDFFN project 1024 -> 256 (one image): U2 = U1 + g @ Wp^T + bp
__global__ __launch_bounds__(128) void k_gemm_wp(const _Float16* __restrict__ A,
                                                 const _Float16* __restrict__ W,
                                                 const float* __restrict__ bias,
                                                 const float* __restrict__ U1,
                                                 float* U2) {
  const int lane = threadIdx.x & 31, wave = threadIdx.x >> 5;
  const int m0 = blockIdx.x << 4, n0 = wave << 6;
  v8f acc[4] = {};
  gemm4(A, CH, W, CH, m0, n0, CH, lane, acc);
  EPILOGUE_BEGIN()
    const size_t i = (size_t)row * CDIM + col;
    U2[i] = U1[i] + v + bias[col];
  EPILOGUE_END()
}

// out_proj 256 -> 64, SiLU, store bohw. One wave per 16x16 tile.
__global__ __launch_bounds__(128) void k_gemm_out(const _Float16* __restrict__ A,
                                                  const _Float16* __restrict__ W,
                                                  float* out) {
  const int lane = threadIdx.x & 31, wave = threadIdx.x >> 5;
  const int m0 = blockIdx.x << 4, n0 = wave << 4;
  v8f acc = {};
  for (int k0 = 0; k0 < CDIM; k0 += 32) {
    v16h a = load_a_frag(A, CDIM, m0, k0, lane);
    v16h b = load_b_frag(W, CDIM, n0, k0, lane);
    acc = wmma_f16(a, b, acc);
  }
  const int half = lane >> 4, nl = lane & 15;
#pragma unroll
  for (int r = 0; r < 8; ++r) {
    const int row = m0 + r + (half << 3);
    const int o = n0 + nl;
    const int b = row >> 12, hw = row & 4095;
    const float v = acc[r];
    out[(size_t)((b << 6) + o) * HW + hw] = v / (1.f + __expf(-v));
  }
}

// ------------------------------- launcher ----------------------------------

extern "C" void kernel_launch(void* const* d_in, const int* in_sizes, int n_in,
                              void* d_out, int out_size, void* d_ws, size_t ws_size,
                              hipStream_t stream) {
  const float* x        = (const float*)d_in[0];
  const float* in_w     = (const float*)d_in[1];
  const float* wq_w     = (const float*)d_in[2];
  const float* wq_b     = (const float*)d_in[3];
  const float* wk_w     = (const float*)d_in[4];
  const float* wk_b     = (const float*)d_in[5];
  const float* wv_w     = (const float*)d_in[6];
  const float* wv_b     = (const float*)d_in[7];
  const float* wo_w     = (const float*)d_in[8];
  const float* wo_b     = (const float*)d_in[9];
  const float* psi_w    = (const float*)d_in[10];
  const float* psi_b    = (const float*)d_in[11];
  const float* dyt_g    = (const float*)d_in[12];
  const float* dyt_b    = (const float*)d_in[13];
  const float* m1_w     = (const float*)d_in[14];
  const float* m1_b     = (const float*)d_in[15];
  const float* m2_w     = (const float*)d_in[16];
  const float* m2_b     = (const float*)d_in[17];
  const float* we_w     = (const float*)d_in[18];
  const float* we_b     = (const float*)d_in[19];
  const float* dw_w     = (const float*)d_in[20];
  const float* dw_b     = (const float*)d_in[21];
  const float* wp_w     = (const float*)d_in[22];
  const float* wp_b     = (const float*)d_in[23];
  const float* op_w     = (const float*)d_in[24];
  float* outp           = (float*)d_out;

  // ---- workspace arena (lifetime-based aliasing) ----
  char* Wb = (char*)d_ws;
  size_t off = 0;
  auto alloc = [&](size_t bytes) -> void* {
    void* p = Wb + off;
    off += (bytes + 255) & ~(size_t)255;
    return p;
  };
  float*    posf  = (float*)   alloc((size_t)HW * CDIM * 4);
  float*    h32   = (float*)   alloc((size_t)NTOK * CDIM * 4);  // later: U2/D2
  _Float16* h16   = (_Float16*)alloc((size_t)NTOK * CDIM * 2);  // later: attn
  float*    psi   = (float*)   alloc((size_t)NTOK * CDIM * 4);  // later: a/D1, then hE
  _Float16* phiQ  = (_Float16*)alloc((size_t)NTOK * CDIM * 2);  // later: U1 f16
  _Float16* phiKT = (_Float16*)alloc((size_t)NTOK * CDIM * 2);  // later: Y f16
  _Float16* VT    = (_Float16*)alloc((size_t)NTOK * CDIM * 2);  // later: gelu f16
  _Float16* kvT   = (_Float16*)alloc((size_t)BATCH * CDIM * CDIM * 2);
  float*    ksum  = (float*)   alloc((size_t)BATCH * CDIM * 4);
  float*    den   = (float*)   alloc((size_t)NTOK * 4);
  float*    pool1 = (float*)   alloc((size_t)BATCH * CDIM * 4);
  float*    s1    = (float*)   alloc((size_t)BATCH * CDIM * 4);
  float*    pool2 = (float*)   alloc((size_t)BATCH * CDIM * 4);
  float*    s2    = (float*)   alloc((size_t)BATCH * CDIM * 4);
  float*    U1f   = (float*)   alloc((size_t)NTOK * CDIM * 4);
  _Float16* wq16  = (_Float16*)alloc((size_t)CDIM * CDIM * 2);
  _Float16* wk16  = (_Float16*)alloc((size_t)CDIM * CDIM * 2);
  _Float16* wv16  = (_Float16*)alloc((size_t)CDIM * CDIM * 2);
  _Float16* wo16  = (_Float16*)alloc((size_t)CDIM * CDIM * 2);
  _Float16* we16  = (_Float16*)alloc((size_t)CH * CDIM * 2);
  _Float16* wp16  = (_Float16*)alloc((size_t)CDIM * CH * 2);
  _Float16* op16  = (_Float16*)alloc((size_t)64 * CDIM * 2);
  if (off > ws_size) return;  // insufficient workspace; bail safely

  // aliases (non-overlapping in time, see lifetime notes above)
  float*    a32  = psi;                 // O-proj output (psi dead)
  float*    D1   = psi;                 // DyT1 output, in-place over a32
  float*    hEimg= psi;                 // per-image EDFFN hidden (D1 dead)
  _Float16* attn = h16;                 // attn (h16 dead after V GEMM)
  _Float16* U1h  = phiQ;                // U1 f16 (phiQ dead after num/den)
  _Float16* Y16  = phiKT;               // final gated f16 (phiKT dead)
  _Float16* g16  = VT;                  // per-image gelu f16 (VT dead)
  float*    U2   = h32;                 // U2 over h32 (h32 dead after DyT1)

  // ---- weight f32 -> f16 conversions ----
  k_cvt<<<(CDIM * CDIM + 255) / 256, 256, 0, stream>>>(wq_w, wq16, CDIM * CDIM);
  k_cvt<<<(CDIM * CDIM + 255) / 256, 256, 0, stream>>>(wk_w, wk16, CDIM * CDIM);
  k_cvt<<<(CDIM * CDIM + 255) / 256, 256, 0, stream>>>(wv_w, wv16, CDIM * CDIM);
  k_cvt<<<(CDIM * CDIM + 255) / 256, 256, 0, stream>>>(wo_w, wo16, CDIM * CDIM);
  k_cvt<<<(CH * CDIM + 255) / 256, 256, 0, stream>>>(we_w, we16, CH * CDIM);
  k_cvt<<<(CDIM * CH + 255) / 256, 256, 0, stream>>>(wp_w, wp16, CDIM * CH);
  k_cvt<<<(64 * CDIM + 255) / 256, 256, 0, stream>>>(op_w, op16, 64 * CDIM);

  // ---- input projection + positional embedding + psi gate ----
  k_pos<<<HW, 256, 0, stream>>>(posf);
  k_inproj<<<NTOK, 256, 0, stream>>>(x, in_w, h32, h16);
  k_psi<<<NTOK, 256, 0, stream>>>(h32, psi_w, psi_b, psi);

  // ---- SAPA linear attention ----
  k_gemm_phi<<<NTOK / 16, 128, 0, stream>>>(h16, wq16, wq_b, posf, psi, phiQ, 0);
  k_gemm_phi<<<NTOK / 16, 128, 0, stream>>>(h16, wk16, wk_b, posf, psi, phiKT, 1);
  k_gemm_v  <<<NTOK / 16, 128, 0, stream>>>(h16, wv16, wv_b, VT);
  k_ksum<<<(BATCH * CDIM) / 8, 256, 0, stream>>>(phiKT, ksum);
  k_gemm_kv<<<dim3(CDIM / 16, BATCH), 128, 0, stream>>>(VT, phiKT, kvT);
  k_den<<<NTOK / 8, 256, 0, stream>>>(phiQ, ksum, den);
  k_gemm_num<<<NTOK / 16, 128, 0, stream>>>(phiQ, kvT, den, attn);
  k_gemm_o<<<NTOK / 16, 128, 0, stream>>>(attn, wo16, wo_b, a32);

  // ---- U1 = Mona(DyT(h + a)) ----
  hipMemsetAsync(pool1, 0, (size_t)BATCH * CDIM * 4, stream);
  k_dyt_pool<<<NTOK, 256, 0, stream>>>(h32, a32, dyt_g, dyt_b, D1, pool1);
  k_mona<<<BATCH, 256, 0, stream>>>(pool1, m1_w, m1_b, s1);
  k_gate<<<NTOK, 256, 0, stream>>>(D1, s1, U1f, U1h);

  // ---- EDFFN, chunked per image to bound the 1024-wide buffers ----
  for (int b = 0; b < BATCH; ++b) {
    const size_t toff = (size_t)b * HW * CDIM;
    k_gemm_we<<<dim3(HW / 16, CH / 256), 128, 0, stream>>>(U1h + toff, we16, we_b, hEimg);
    k_dwgelu<<<HW, 256, 0, stream>>>(hEimg, dw_w, dw_b, g16);
    k_gemm_wp<<<HW / 16, 128, 0, stream>>>(g16, wp16, wp_b, U1f + toff, U2 + toff);
  }

  // ---- Y = Mona(DyT(U2)) ; out = silu(Y @ out_proj^T) in bohw ----
  hipMemsetAsync(pool2, 0, (size_t)BATCH * CDIM * 4, stream);
  k_dyt_pool<<<NTOK, 256, 0, stream>>>(U2, nullptr, dyt_g, dyt_b, U2, pool2);
  k_mona<<<BATCH, 256, 0, stream>>>(pool2, m2_w, m2_b, s2);
  k_gate<<<NTOK, 256, 0, stream>>>(U2, s2, nullptr, Y16);
  k_gemm_out<<<NTOK / 16, 128, 0, stream>>>(Y16, op16, outp);
}
